// ModulatedConv2dClean_3968549781778
// MI455X (gfx1250) — matrix operand order repaired and checked by reading, not compiled, not run on hardware
//
#include <hip/hip_runtime.h>
#include <hip/hip_bf16.h>

typedef __attribute__((ext_vector_type(16))) __bf16 v16bf;
typedef __attribute__((ext_vector_type(8)))  __bf16 v8bf;
typedef __attribute__((ext_vector_type(8)))  float  v8f;

#define B_    16
#define CIN   64
#define COUT  64
#define S_DIM 512
#define H_    256
#define W_    256
#define KK    9
#define KTOT  (CIN * KK)   // 576 = 18 * 32
#define NTILE 32
#define EPSF  1e-8f

// ---------------------------------------------------------------------------
// Stage 1: s[b][c] = dot(style[b], mod_weight[c]) + mod_bias[c]
// ---------------------------------------------------------------------------
__global__ void style_kernel(const float* __restrict__ style,
                             const float* __restrict__ mw,
                             const float* __restrict__ mb,
                             float* __restrict__ s_out) {
    const int b = blockIdx.x;
    const int c = threadIdx.x;
    const float* st = style + (size_t)b * S_DIM;
    const float* wr = mw + (size_t)c * S_DIM;
    float acc = mb[c];
#pragma unroll 4
    for (int j = 0; j < S_DIM; ++j) acc += st[j] * wr[j];
    s_out[b * CIN + c] = acc;
}

// ---------------------------------------------------------------------------
// Stage 2: modulate + demodulate, emit bf16 weight matrix
//          wmat[b][o][k]  with k = c*9 + kh*3 + kw   (row-major, K contiguous)
// ---------------------------------------------------------------------------
__global__ void weight_kernel(const float* __restrict__ weight,  // (1,O,C,3,3)
                              const float* __restrict__ s,       // (B,C)
                              __bf16* __restrict__ wmat) {
    const int bo = blockIdx.x;            // b*COUT + o
    const int b  = bo / COUT;
    const int o  = bo - b * COUT;
    const int c  = threadIdx.x;           // 64 threads, one per input channel

    const float sc = s[b * CIN + c];
    const float* wp = weight + ((size_t)o * CIN + c) * KK;

    float wl[KK];
    float ss = 0.f;
#pragma unroll
    for (int i = 0; i < KK; ++i) {
        float v = wp[i] * sc;
        wl[i] = v;
        ss += v * v;
    }

    __shared__ float red[CIN];
    red[c] = ss;
    __syncthreads();
    for (int ofs = CIN / 2; ofs > 0; ofs >>= 1) {
        if (c < ofs) red[c] += red[c + ofs];
        __syncthreads();
    }
    const float demod = rsqrtf(red[0] + EPSF);

    __bf16* dst = wmat + (size_t)bo * KTOT + c * KK;
#pragma unroll
    for (int i = 0; i < KK; ++i) dst[i] = (__bf16)(wl[i] * demod);
}

// ---------------------------------------------------------------------------
// Stage 3: implicit-GEMM conv via v_wmma_f32_16x16x32_bf16
//   grid = (2048 spatial tiles, 16 batch), block = 128 threads (4 waves)
//   Each workgroup: out[b][0..63][h][w0..w0+31]   (M=64, N=32, K=576)
// ---------------------------------------------------------------------------
__global__ __launch_bounds__(128)
void conv_kernel(const float* __restrict__ x,
                 const __bf16* __restrict__ wmat,
                 float* __restrict__ out) {
    __shared__ __bf16 Bs[NTILE][KTOT];    // im2col panel [n][k], 36,864 bytes

    const int tid  = threadIdx.x;
    const int lane = tid & 31;
    const int wave = tid >> 5;            // M block: rows wave*16 .. wave*16+15
    const int tileIdx = blockIdx.x;       // 0..2047
    const int b  = blockIdx.y;            // 0..15
    const int h  = tileIdx >> 3;          // 8 column tiles per image row
    const int w0 = (tileIdx & 7) * NTILE;

    const float*  xb   = x + (size_t)b * CIN * H_ * W_;
    const __bf16* arowB = wmat + (size_t)b * COUT * KTOT;

    // prefetch this sample's A panel (global_prefetch_b8)
    if (tid < COUT) __builtin_prefetch(arowB + (size_t)tid * KTOT, 0, 0);

    // ---- build im2col B panel in LDS (bf16), zero-padded borders ----
    for (int e = tid; e < KTOT; e += 128) {
        const int c  = e / 9;
        const int r  = e - c * 9;
        const int kh = r / 3;
        const int kw = r - kh * 3;
        const int y  = h + kh - 1;
        const bool yok = (unsigned)y < (unsigned)H_;
        const float* row = xb + ((size_t)c * H_ + y) * W_;
#pragma unroll 8
        for (int n = 0; n < NTILE; ++n) {
            const int xx = w0 + n + kw - 1;
            const float v = (yok && (unsigned)xx < (unsigned)W_) ? row[xx] : 0.f;
            Bs[n][e] = (__bf16)v;
        }
    }
    __syncthreads();

    // ---- WMMA main loop ----
    v8f acc0 = {};
    v8f acc1 = {};
    const int rowM = wave * 16 + (lane & 15);
    const int kb_a = (lane >> 4) * 16;    // A: lanes 16-31 hold K+16 chunk
    const int kb_b = (lane >> 4) * 16;    // B: lanes 16-31 hold K+16 chunk
    const int nb0  = (lane & 15);
    const int nb1  = 16 + (lane & 15);
    const __bf16* arow = arowB + (size_t)rowM * KTOT;

#pragma unroll 3
    for (int k0 = 0; k0 < KTOT; k0 += 32) {
        union { v16bf v; v8bf h[2]; } A, Bf0, Bf1;
        A.h[0]   = *(const v8bf*)(arow + k0 + kb_a);
        A.h[1]   = *(const v8bf*)(arow + k0 + kb_a + 8);
        Bf0.h[0] = *(const v8bf*)(&Bs[nb0][k0 + kb_b]);
        Bf0.h[1] = *(const v8bf*)(&Bs[nb0][k0 + kb_b + 8]);
        Bf1.h[0] = *(const v8bf*)(&Bs[nb1][k0 + kb_b]);
        Bf1.h[1] = *(const v8bf*)(&Bs[nb1][k0 + kb_b + 8]);
        acc0 = __builtin_amdgcn_wmma_f32_16x16x32_bf16(
                   false, A.v, false, Bf0.v, (short)0, acc0, false, false);
        acc1 = __builtin_amdgcn_wmma_f32_16x16x32_bf16(
                   false, A.v, false, Bf1.v, (short)0, acc1, false, false);
    }

    // ---- store D: vgpr v -> row (v + (lane>>4)*8), col (lane&15) ----
    const int oBase = wave * 16 + ((lane >> 4) << 3);
    float* outp = out + (((size_t)b * COUT) * H_ + h) * W_ + w0;
#pragma unroll
    for (int v = 0; v < 8; ++v) {
        float* orow = outp + (size_t)(oBase + v) * H_ * W_;
        orow[nb0] = acc0[v];
        orow[nb1] = acc1[v];
    }
}

// ---------------------------------------------------------------------------
extern "C" void kernel_launch(void* const* d_in, const int* in_sizes, int n_in,
                              void* d_out, int out_size, void* d_ws, size_t ws_size,
                              hipStream_t stream) {
    const float* x      = (const float*)d_in[0];   // (16,64,256,256)
    const float* style  = (const float*)d_in[1];   // (16,512)
    const float* mw     = (const float*)d_in[2];   // (64,512)
    const float* mb     = (const float*)d_in[3];   // (64,)
    const float* weight = (const float*)d_in[4];   // (1,64,64,3,3)
    float* out = (float*)d_out;

    float*  s_ws = (float*)d_ws;                             // 4 KB
    __bf16* wmat = (__bf16*)((char*)d_ws + 4096);            // 16*64*576 bf16 ~1.1 MB

    style_kernel <<<B_, CIN, 0, stream>>>(style, mw, mb, s_ws);
    weight_kernel<<<B_ * COUT, CIN, 0, stream>>>(weight, s_ws, wmat);

    dim3 grid((H_ * W_) / NTILE / 1, B_);   // 2048 x 16
    conv_kernel<<<grid, 128, 0, stream>>>(x, wmat, out);
}
